// KANLayer_23957327577247
// MI455X (gfx1250) — matrix operand order repaired and checked by reading, not compiled, not run on hardware
//
#include <hip/hip_runtime.h>

// ---------------------------------------------------------------------------
// KAN layer as one f16 WMMA GEMM on gfx1250 (wave32).
//
// Augmented-K: K' = 512 features * 8 slots = 4096
//   slot 0 -> raw x, slots 1..5 -> normalized Gaussian basis, slots 6,7 -> 0
// out[32768,512] = A[32768,4096] @ W16[4096,512]
//
// Roofline: 137 GFLOP of f16 WMMA vs ~132 MB HBM (~5.7us @ 23.3 TB/s):
// matrix math is cheap; the watchpoints are (a) basis VALU work and (b)
// staging. Basis uses the uniform-grid factorization
//   exp(-10(x-g_j)^2) = exp(-10x^2) * exp(20*g0*x) * u^j * exp(-10*g_j^2),
//   u = exp(20*h*x)
// (exp(-10x^2) cancels in normalization; sum(basis) >= 0.85 so the 1e-8
// epsilon is negligible) -> 2 exps/pair instead of 5. Tiles are 128x256 so
// the basis is computed only 2x total. LDS double-buffered, 1 barrier/step;
// B staging via GLOBAL_LOAD_ASYNC_TO_LDS_B128 (ASYNCcnt-tracked).
// ---------------------------------------------------------------------------

typedef __attribute__((ext_vector_type(16))) _Float16 v16h;
typedef __attribute__((ext_vector_type(8)))  _Float16 v8h;
typedef __attribute__((ext_vector_type(8)))  float    v8f;
typedef __attribute__((ext_vector_type(4)))  int      v4i;

#define KAN_IN   512
#define KAN_OUT  512
#define KAN_GRID 5
#define KSLOT    8
#define KTOT     (KAN_IN * KSLOT)     // 4096

#if defined(__HIP_DEVICE_COMPILE__) && __has_builtin(__builtin_amdgcn_global_load_async_to_lds_b128)
#define KAN_ASYNC 1
#if __has_builtin(__builtin_amdgcn_s_wait_asynccnt)
#define KAN_WAIT_ASYNC() __builtin_amdgcn_s_wait_asynccnt(0)
#else
#define KAN_WAIT_ASYNC() asm volatile("s_wait_asynccnt 0x0" ::: "memory")
#endif
#endif

// ---------------------------------------------------------------------------
// Pack weights into f16 [K'=4096, OUT=512], pre-swizzled into WMMA B-fragment
// order. Chunk = (cb*128 + kchunk32), cb in {0,1} = 256-col block:
//   flat = (chunk*16 + colTile)*512 + lane*16 + e
// B 32x16 fragment, this code's contiguous k order (A matches): lane = N + 16*hi, K = e + 16*hi.
// ---------------------------------------------------------------------------
__global__ __launch_bounds__(256) void kan_pack_w(
    const float* __restrict__ base_w,    // [OUT, IN]
    const float* __restrict__ spline_w,  // [OUT, IN, GRID]
    _Float16*    __restrict__ Bmat)      // 2M f16 (4 MB)
{
  int idx  = blockIdx.x * blockDim.x + threadIdx.x;   // 0 .. 2M-1
  int e    = idx & 15;
  int lane = (idx >> 4) & 31;
  int ct   = (idx >> 9) & 15;    // 16-col tile within 256-col block
  int kk   = (idx >> 13) & 127;  // 32-K chunk
  int cb   = (idx >> 20) & 1;    // 256-col block

  int k = (kk << 5) + e + ((lane >> 4) << 4);     // global K'
  int o = (cb << 8) + (ct << 4) + (lane & 15);
  int i = k >> 3;
  int s = k & 7;

  float v = 0.0f;
  if (s == 0)             v = base_w[o * KAN_IN + i];
  else if (s <= KAN_GRID) v = spline_w[(o * KAN_IN + i) * KAN_GRID + (s - 1)];
  Bmat[idx] = (_Float16)v;
}

// ---------------------------------------------------------------------------
// GEMM: block = 256 threads (8 waves), output tile 128 rows x 256 cols.
// Wave grid 2x4 -> wave tile 64x64 = 4x4 WMMA accumulators.
// K in steps of 64 (8 input features), double-buffered LDS, 1 barrier/step.
// ---------------------------------------------------------------------------
__global__ __launch_bounds__(256) void kan_wmma_gemm(
    const float*    __restrict__ x,     // [rows, 512]
    const _Float16* __restrict__ Bmat,  // swizzled
    const float*    __restrict__ grid,  // [5]
    float*          __restrict__ out,   // [rows, 512]
    int rows)
{
  __shared__ __align__(16) _Float16 sA[2 * 8192];   // 2 bufs x (128x64)  32 KB
  __shared__ __align__(16) _Float16 sB[2 * 16384];  // 2 bufs x (64x256)  64 KB

  const int tid     = threadIdx.x;
  const int lane    = tid & 31;
  const int wave    = tid >> 5;
  const int wr      = wave >> 2;          // 0..1
  const int wc      = wave & 3;           // 0..3
  const int cb      = blockIdx.x;         // 0..1 (256-col block)
  const int rowBase = blockIdx.y << 7;

  // Basis constants (hoisted): assumes uniform grid (reference: linspace).
  const float g0 = grid[0], g1 = grid[1], g2 = grid[2], g3 = grid[3], g4 = grid[4];
  const float h   = g1 - g0;
  const float a20 = 20.0f * h;
  const float b20 = 20.0f * g0;
  const float c0 = __expf(-10.0f * g0 * g0);
  const float c1 = __expf(-10.0f * g1 * g1);
  const float c2 = __expf(-10.0f * g2 * g2);
  const float c3 = __expf(-10.0f * g3 * g3);
  const float c4 = __expf(-10.0f * g4 * g4);

  // Stage the A chunk for step kkS into LDS buffer bufS (A-fragment swizzle:
  // for k_local = ii*8+s: frag-half = ii>>2, lane = M + 16*((ii>>1)&1),
  // elem e = s + 8*(ii&1) -> one contiguous v8h store per (row,feature)).
  auto stageA = [&](int kkS, int bufS) {
    #pragma unroll
    for (int rep = 0; rep < 4; ++rep) {
      int p  = tid + (rep << 8);          // 0..1023
      int r  = p >> 3;
      int ii = p & 7;
      float xv = x[(size_t)(rowBase + r) * KAN_IN + (kkS << 3) + ii];
      float xc = fminf(fmaxf((xv + 1.0f) * 0.5f, 0.0f), 1.0f);
      float u  = __expf(a20 * xc);
      float w  = __expf(b20 * xc);
      float t0 = w * c0;
      float m  = w * u;
      float t1 = m * c1;  m *= u;
      float t2 = m * c2;  m *= u;
      float t3 = m * c3;  m *= u;
      float t4 = m * c4;
      float inv = __builtin_amdgcn_rcpf(t0 + t1 + t2 + t3 + t4);
      v8h pk;
      pk[0] = (_Float16)xv;
      pk[1] = (_Float16)(t0 * inv);
      pk[2] = (_Float16)(t1 * inv);
      pk[3] = (_Float16)(t2 * inv);
      pk[4] = (_Float16)(t3 * inv);
      pk[5] = (_Float16)(t4 * inv);
      pk[6] = (_Float16)0.0f;
      pk[7] = (_Float16)0.0f;
      _Float16* dst = sA + (bufS << 13) + ((ii >> 2) << 12) + ((r >> 4) << 9)
                    + (((r & 15) + (((ii >> 1) & 1) << 4)) << 4) + ((ii & 1) << 3);
      *(v8h*)dst = pk;
    }
  };

  // Stage the (pre-swizzled) B chunks for step kkS: pure contiguous copy.
  auto stageB = [&](int kkS, int bufS) {
    #pragma unroll
    for (int ks = 0; ks < 2; ++ks) {
      const _Float16* src =
          Bmat + (((size_t)(cb << 7) + (kkS << 1) + ks) << 13) + (tid << 5);
      _Float16* dst = sB + (bufS << 14) + (ks << 13) + (tid << 5);
#ifdef KAN_ASYNC
      #pragma unroll
      for (int q = 0; q < 4; ++q)
        __builtin_amdgcn_global_load_async_to_lds_b128(
            (__attribute__((address_space(1))) v4i*)(src + q * 8),
            (__attribute__((address_space(3))) v4i*)(dst + q * 8), 0, 0);
#else
      #pragma unroll
      for (int q = 0; q < 4; ++q)
        *(v8h*)(dst + q * 8) = *(const v8h*)(src + q * 8);
#endif
    }
  };

  v8f c[4][4];
  #pragma unroll
  for (int rt = 0; rt < 4; ++rt)
    #pragma unroll
    for (int ct = 0; ct < 4; ++ct)
      c[rt][ct] = v8f{0.f, 0.f, 0.f, 0.f, 0.f, 0.f, 0.f, 0.f};

  // Prologue: stage step 0 into buffer 0.
  stageA(0, 0);
  stageB(0, 0);

  for (int kk = 0; kk < 64; ++kk) {
    const int cbuf = kk & 1;
#ifdef KAN_ASYNC
    KAN_WAIT_ASYNC();          // async B for buffer cbuf has landed in LDS
#endif
    __syncthreads();           // emits s_wait_dscnt 0 first -> A stores + prior
                               // reads of buffer cbuf^1 are done in all waves
    if (kk + 1 < 64) {         // stage next step; overlaps with WMMA below
      stageA(kk + 1, cbuf ^ 1);
      stageB(kk + 1, cbuf ^ 1);
    }
#ifndef KAN_ASYNC
    if (kk + 2 < 64)
      __builtin_prefetch(
          Bmat + (((size_t)(cb << 7) + ((kk + 2) << 1)) << 13) + (tid << 5), 0, 0);
#endif

    #pragma unroll
    for (int ks = 0; ks < 2; ++ks) {
      v16h a[4], b[4];
      #pragma unroll
      for (int rt = 0; rt < 4; ++rt)
        a[rt] = *(const v16h*)(sA + (cbuf << 13) + (ks << 12)
                               + (((wr << 2) + rt) << 9) + (lane << 4));
      #pragma unroll
      for (int ct = 0; ct < 4; ++ct)
        b[ct] = *(const v16h*)(sB + (cbuf << 14) + (ks << 13)
                               + (((wc << 2) + ct) << 9) + (lane << 4));
      #pragma unroll
      for (int rt = 0; rt < 4; ++rt)
        #pragma unroll
        for (int ct = 0; ct < 4; ++ct)
          c[rt][ct] = __builtin_amdgcn_wmma_f32_16x16x32_f16(
              false, a[rt], false, b[ct], (short)0, c[rt][ct], false, false);
    }
  }

  // Epilogue: C/D layout: VGPR r holds M = r + 8*(lane>=16), N = lane&15.
  const int hi16 = lane >> 4;
  const int nl   = lane & 15;
  #pragma unroll
  for (int rt = 0; rt < 4; ++rt) {
    int m0 = rowBase + (wr << 6) + (rt << 4) + (hi16 << 3);
    #pragma unroll
    for (int ct = 0; ct < 4; ++ct) {
      int o0 = (cb << 8) + (wc << 6) + (ct << 4) + nl;
      #pragma unroll
      for (int r = 0; r < 8; ++r)
        out[(size_t)(m0 + r) * KAN_OUT + o0] = c[rt][ct][r];
    }
  }
}

// ---------------------------------------------------------------------------
extern "C" void kernel_launch(void* const* d_in, const int* in_sizes, int n_in,
                              void* d_out, int out_size, void* d_ws, size_t ws_size,
                              hipStream_t stream) {
  const float* x        = (const float*)d_in[0];
  const float* base_w   = (const float*)d_in[1];
  const float* spline_w = (const float*)d_in[2];
  const float* grid     = (const float*)d_in[3];
  float*       out      = (float*)d_out;
  _Float16*    Bmat     = (_Float16*)d_ws;        // 4 MB packed weights

  const int rows = in_sizes[0] / KAN_IN;          // 32768

  kan_pack_w<<<(KAN_OUT * KTOT) / 256, 256, 0, stream>>>(base_w, spline_w, Bmat);

  dim3 gridDim(KAN_OUT / 256, rows / 128);        // 2 x 256 blocks
  kan_wmma_gemm<<<gridDim, 256, 0, stream>>>(x, Bmat, grid, out, rows);
}
